// DecoderRNN_21577915695142
// MI455X (gfx1250) — compile-verified, matrix operand order checked by reading, not run on hardware
//
#include <hip/hip_runtime.h>
#include <hip/hip_bf16.h>
#include <math.h>

// ---------------------------------------------------------------------------
// Sizes (from reference setup_inputs)
// ---------------------------------------------------------------------------
#define BATCH   512
#define HID     1024
#define VOCAB   512
#define TH3     3072          // 3*HID
#define SEQL    120
#define CIN     128
#define OCH     32
#define KW      10
#define LOUT    111           // SEQL-KW+1
#define KCONV   1280          // KW*CIN contraction depth
#define PADLEN  30
#define NKT_H   32            // HID/32 k-tiles
#define NKT_C   40            // KCONV/32 k-tiles

typedef __attribute__((ext_vector_type(16))) __bf16 v16bf;
typedef __attribute__((ext_vector_type(8)))  __bf16 v8bf;
typedef __attribute__((ext_vector_type(8)))  float  v8f;

// ---------------------------------------------------------------------------
// WMMA helpers (CDNA5 16x16x32 bf16, f32 accumulate)
// ---------------------------------------------------------------------------
__device__ __forceinline__ v8f wmma_bf16(v16bf a, v16bf b, v8f c) {
    return __builtin_amdgcn_wmma_f32_16x16x32_bf16(
        /*neg_a=*/false, a, /*neg_b=*/false, b,
        /*c_mod=*/(short)0, c, /*reuse_a=*/false, /*reuse_b=*/false);
}

__device__ __forceinline__ v16bf cat8(v8bf lo, v8bf hi) {
    return __builtin_shufflevector(lo, hi, 0,1,2,3,4,5,6,7,8,9,10,11,12,13,14,15);
}

// A-operand fragment for 16-bit 16x32: lane holds row M=lane&15.
// K mapping per ISA: lanes<16  -> e<8: K=e,     e>=8: K=e+8
//                    lanes>=16 -> e<8: K=e+8,   e>=8: K=e+16
// => two contiguous 8-element (16B) chunks at offsets koff and 16+koff.
__device__ __forceinline__ v16bf load_a_frag(const __bf16* rowk, int koff) {
    v8bf lo = *(const v8bf*)(rowk + koff);
    v8bf hi = *(const v8bf*)(rowk + 16 + koff);
    return cat8(lo, hi);
}

__device__ __forceinline__ float sigmoidf_(float x) {
    return 1.0f / (1.0f + __expf(-x));
}

// ---------------------------------------------------------------------------
// Pack kernels: fp32 weight [N,K] row-major -> per-lane contiguous bf16
// B-operand fragments, layout [nt][kt][lane][16]
// ---------------------------------------------------------------------------
__global__ void pack_weight_b(const float* __restrict__ W, __bf16* __restrict__ out,
                              int N, int K) {
    int nkt = K >> 5;
    int tid = blockIdx.x * blockDim.x + threadIdx.x;
    int total = (N >> 4) * nkt * 32;
    if (tid >= total) return;
    int lane = tid & 31;
    int frag = tid >> 5;
    int kt = frag % nkt;
    int nt = frag / nkt;
    int n = nt * 16 + (lane & 15);
    int koff = (lane >= 16) ? 8 : 0;
    __bf16* dst = out + (size_t)tid * 16;
#pragma unroll
    for (int e = 0; e < 16; ++e) {
        int k = kt * 32 + ((e < 8) ? e : e + 8) + koff;
        dst[e] = (__bf16)W[(size_t)n * K + k];
    }
}

// conv_w [O,C,KW]: contraction index j = kpos*CIN + c  (matches A layout of
// sm_hidden flattened [B, L*C]); B[j][o] = conv_w[o][c][kpos]
__global__ void pack_conv_b(const float* __restrict__ cw, __bf16* __restrict__ out) {
    int tid = blockIdx.x * blockDim.x + threadIdx.x;
    int total = (OCH / 16) * NKT_C * 32;
    if (tid >= total) return;
    int lane = tid & 31;
    int frag = tid >> 5;
    int kt = frag % NKT_C;
    int nt = frag / NKT_C;
    int o = nt * 16 + (lane & 15);
    int koff = (lane >= 16) ? 8 : 0;
    __bf16* dst = out + (size_t)tid * 16;
#pragma unroll
    for (int e = 0; e < 16; ++e) {
        int j = kt * 32 + ((e < 8) ? e : e + 8) + koff;
        int kpos = j >> 7;         // j / CIN
        int c    = j & 127;        // j % CIN
        dst[e] = (__bf16)cw[(size_t)o * CIN * KW + (size_t)c * KW + kpos];
    }
}

__global__ void cvt_bf16_kernel(const float* __restrict__ in, __bf16* __restrict__ out, int n) {
    int i = blockIdx.x * blockDim.x + threadIdx.x;
    if (i < n) out[i] = (__bf16)in[i];
}

// ---------------------------------------------------------------------------
// Encoder: conv1d(K=10) as 1280-deep GEMM per position + relu + time-maxpool,
// then [B,32] @ enc_w.T -> relu -> h0 [B,1024].  One block per 16 rows.
// ---------------------------------------------------------------------------
__global__ __launch_bounds__(256) void encoder_kernel(
    const __bf16* __restrict__ smb,      // [B][SEQL*CIN] bf16
    const __bf16* __restrict__ convb,    // packed [2][40][32][16]
    const float*  __restrict__ conv_bias,
    const __bf16* __restrict__ encwb,    // packed [64][1][32][16]
    const float*  __restrict__ enc_bias,
    float* __restrict__ h0)
{
    __shared__ unsigned pooled_bits[16 * OCH];   // relu>=0 -> uint-ordered max
    const int w    = threadIdx.x >> 5;
    const int lane = threadIdx.x & 31;
    const int b0   = blockIdx.x * 16;
    const int m    = lane & 15;
    const int mrow = (lane >= 16) ? 8 : 0;
    const int koff = (lane >= 16) ? 8 : 0;

    pooled_bits[threadIdx.x]       = 0u;
    pooled_bits[threadIdx.x + 256] = 0u;
    __syncthreads();

    float mx0[8], mx1[8];
#pragma unroll
    for (int f = 0; f < 8; ++f) { mx0[f] = 0.f; mx1[f] = 0.f; }

    const __bf16* arow = smb + (size_t)(b0 + m) * (SEQL * CIN);
    const float cb0 = conv_bias[m];
    const float cb1 = conv_bias[16 + m];

    for (int t = w; t < LOUT; t += 8) {
        v8f acc0 = {0,0,0,0,0,0,0,0};
        v8f acc1 = {0,0,0,0,0,0,0,0};
        for (int kt = 0; kt < NKT_C; ++kt) {
            v16bf a  = load_a_frag(arow + t * CIN + kt * 32, koff);
            v16bf bA = *(const v16bf*)(convb + ((size_t)(0 * NKT_C + kt) * 32 + lane) * 16);
            v16bf bB = *(const v16bf*)(convb + ((size_t)(1 * NKT_C + kt) * 32 + lane) * 16);
            acc0 = wmma_bf16(a, bA, acc0);
            acc1 = wmma_bf16(a, bB, acc1);
        }
#pragma unroll
        for (int f = 0; f < 8; ++f) {
            mx0[f] = fmaxf(mx0[f], fmaxf(acc0[f] + cb0, 0.f));
            mx1[f] = fmaxf(mx1[f], fmaxf(acc1[f] + cb1, 0.f));
        }
    }
#pragma unroll
    for (int f = 0; f < 8; ++f) {
        atomicMax(&pooled_bits[(f + mrow) * OCH + m],      __float_as_uint(mx0[f]));
        atomicMax(&pooled_bits[(f + mrow) * OCH + 16 + m], __float_as_uint(mx1[f]));
    }
    __syncthreads();

    // encoder linear: K=32 (one k-tile), N=1024 (64 n-tiles, 8 per wave)
    v16bf ap;
#pragma unroll
    for (int e = 0; e < 16; ++e) {
        int k = ((e < 8) ? e : e + 8) + koff;
        ap[e] = (__bf16)__uint_as_float(pooled_bits[m * OCH + k]);
    }
    for (int i = 0; i < 8; ++i) {
        int nt = w * 8 + i;
        v16bf bv = *(const v16bf*)(encwb + ((size_t)nt * 32 + lane) * 16);
        v8f acc = {0,0,0,0,0,0,0,0};
        acc = wmma_bf16(ap, bv, acc);
        int n = nt * 16 + m;
        float eb = enc_bias[n];
#pragma unroll
        for (int f = 0; f < 8; ++f) {
            h0[(size_t)(b0 + f + mrow) * HID + n] = fmaxf(acc[f] + eb, 0.f);
        }
    }
}

// ---------------------------------------------------------------------------
// Persistent greedy GRU decoder: one block per 16 batch rows, 30 steps.
// ---------------------------------------------------------------------------
__global__ __launch_bounds__(256) void decoder_kernel(
    const float*  __restrict__ emb,     // [VOCAB][HID] f32
    const __bf16* __restrict__ wih,     // packed [192][32][32][16]
    const __bf16* __restrict__ whh,     // packed [192][32][32][16]
    const __bf16* __restrict__ wout,    // packed [32][32][32][16]
    const float*  __restrict__ bih,
    const float*  __restrict__ bhh,
    const float*  __restrict__ bout,
    const float*  __restrict__ h0,
    float* __restrict__ out_lsm,        // [B][V]
    float* __restrict__ out_hid,        // [PADLEN][B][HID]
    float* __restrict__ out_idx)        // [PADLEN+1][B]
{
    __shared__ __align__(16) float  sh_h [16 * HID];     // 64KB current h (f32)
    __shared__ __align__(16) __bf16 sh_hb[16 * HID];     // 32KB current h (bf16 A-src)
    __shared__ __align__(16) float  sh_u [16 * VOCAB];   // 32KB union: x(bf16) / logits(f32)
    __shared__ int sh_tok[16];
    __bf16* sh_xb = (__bf16*)sh_u;                       // 16*1024 bf16 == 32KB

    const int w    = threadIdx.x >> 5;
    const int lane = threadIdx.x & 31;
    const int b0   = blockIdx.x * 16;
    const int m    = lane & 15;
    const int mrow = (lane >= 16) ? 8 : 0;
    const int koff = (lane >= 16) ? 8 : 0;

    // init: h = h0, x = embedding[SOS=0], decoded_indices[0] = SOS
    for (int idx = threadIdx.x; idx < 16 * HID; idx += 256) {
        float v = h0[(size_t)b0 * HID + idx];
        sh_h[idx]  = v;
        sh_hb[idx] = (__bf16)v;
        sh_xb[idx] = (__bf16)emb[idx & (HID - 1)];   // row 0 of embedding
    }
    if (threadIdx.x < 16) {
        sh_tok[threadIdx.x] = 0;
        out_idx[b0 + threadIdx.x] = 0.0f;
    }
    __syncthreads();

    for (int s = 0; s < PADLEN; ++s) {
        // ---------------- GRU GEMMs + gates (h_new kept in registers) -------
        float hnew[8][8];
#pragma unroll 1
        for (int jj = 0; jj < 8; ++jj) {
            const int jt = w * 8 + jj;
            v8f air = {0,0,0,0,0,0,0,0}, aiz = air, ain = air;
            v8f ahr = air, ahz = air, ahn = air;
            for (int kt = 0; kt < NKT_H; ++kt) {
                v16bf ax = load_a_frag(sh_xb + m * HID + kt * 32, koff);
                v16bf ah = load_a_frag(sh_hb + m * HID + kt * 32, koff);
                v16bf bir = *(const v16bf*)(wih + (((size_t)(jt      ) * NKT_H + kt) * 32 + lane) * 16);
                v16bf biz = *(const v16bf*)(wih + (((size_t)(jt +  64) * NKT_H + kt) * 32 + lane) * 16);
                v16bf bin = *(const v16bf*)(wih + (((size_t)(jt + 128) * NKT_H + kt) * 32 + lane) * 16);
                v16bf bhr = *(const v16bf*)(whh + (((size_t)(jt      ) * NKT_H + kt) * 32 + lane) * 16);
                v16bf bhz = *(const v16bf*)(whh + (((size_t)(jt +  64) * NKT_H + kt) * 32 + lane) * 16);
                v16bf bhn = *(const v16bf*)(whh + (((size_t)(jt + 128) * NKT_H + kt) * 32 + lane) * 16);
                air = wmma_bf16(ax, bir, air);
                aiz = wmma_bf16(ax, biz, aiz);
                ain = wmma_bf16(ax, bin, ain);
                ahr = wmma_bf16(ah, bhr, ahr);
                ahz = wmma_bf16(ah, bhz, ahz);
                ahn = wmma_bf16(ah, bhn, ahn);
            }
            const int j = jt * 16 + m;
            const float bir_ = bih[j], biz_ = bih[j + HID], bin_ = bih[j + 2 * HID];
            const float bhr_ = bhh[j], bhz_ = bhh[j + HID], bhn_ = bhh[j + 2 * HID];
#pragma unroll
            for (int f = 0; f < 8; ++f) {
                const int mm = f + mrow;
                float r  = sigmoidf_(air[f] + bir_ + ahr[f] + bhr_);
                float z  = sigmoidf_(aiz[f] + biz_ + ahz[f] + bhz_);
                float nn = tanhf(ain[f] + bin_ + r * (ahn[f] + bhn_));
                float ho = sh_h[mm * HID + j];
                float hv = (1.0f - z) * nn + z * ho;
                hnew[jj][f] = hv;
                out_hid[((size_t)s * BATCH + b0 + mm) * HID + j] = hv;
            }
        }
        __syncthreads();   // everyone done reading old h / x
#pragma unroll
        for (int jj = 0; jj < 8; ++jj) {
            const int j = (w * 8 + jj) * 16 + m;
#pragma unroll
            for (int f = 0; f < 8; ++f) {
                const int mm = f + mrow;
                sh_h [mm * HID + j] = hnew[jj][f];
                sh_hb[mm * HID + j] = (__bf16)hnew[jj][f];
            }
        }
        __syncthreads();

        // ---------------- logits = h_new @ w_out.T + b_out ------------------
        float* logits = sh_u;   // overlays sh_xb (x already consumed)
#pragma unroll 1
        for (int i = 0; i < 4; ++i) {
            const int nt = w * 4 + i;
            v8f acc = {0,0,0,0,0,0,0,0};
            for (int kt = 0; kt < NKT_H; ++kt) {
                v16bf ah = load_a_frag(sh_hb + m * HID + kt * 32, koff);
                v16bf bw = *(const v16bf*)(wout + (((size_t)nt * NKT_H + kt) * 32 + lane) * 16);
                acc = wmma_bf16(ah, bw, acc);
            }
            const int n = nt * 16 + m;
            const float bo = bout[n];
#pragma unroll
            for (int f = 0; f < 8; ++f)
                logits[(f + mrow) * VOCAB + n] = acc[f] + bo;
        }
        __syncthreads();

        // ---------------- greedy argmax (+ final log-softmax) --------------
        if (threadIdx.x < 16) {
            const int mm = threadIdx.x;
            const float* lr = logits + mm * VOCAB;
            float best = lr[0]; int bi = 0;
            for (int v = 1; v < VOCAB; ++v) {
                float x = lr[v];
                if (x > best) { best = x; bi = v; }
            }
            sh_tok[mm] = bi;
            out_idx[(size_t)(s + 1) * BATCH + b0 + mm] = (float)bi;
            if (s == PADLEN - 1) {
                float sum = 0.f;
                for (int v = 0; v < VOCAB; ++v) sum += __expf(lr[v] - best);
                float lse = best + __logf(sum);
                for (int v = 0; v < VOCAB; ++v)
                    out_lsm[(size_t)(b0 + mm) * VOCAB + v] = lr[v] - lse;
            }
        }
        __syncthreads();

        // ---------------- stage next x = embedding[token] -------------------
        if (s < PADLEN - 1) {
            for (int idx = threadIdx.x; idx < 16 * HID; idx += 256) {
                int mm = idx >> 10;
                int k  = idx & (HID - 1);
                sh_xb[idx] = (__bf16)emb[(size_t)sh_tok[mm] * HID + k];
            }
        }
        __syncthreads();
    }
}

// ---------------------------------------------------------------------------
// Host launcher
// ---------------------------------------------------------------------------
static inline size_t align256(size_t x) { return (x + 255) & ~(size_t)255; }

extern "C" void kernel_launch(void* const* d_in, const int* in_sizes, int n_in,
                              void* d_out, int out_size, void* d_ws, size_t ws_size,
                              hipStream_t stream) {
    const float* sm_hidden = (const float*)d_in[0];
    const float* embedding = (const float*)d_in[1];
    const float* w_ih      = (const float*)d_in[2];
    const float* w_hh      = (const float*)d_in[3];
    const float* b_ih      = (const float*)d_in[4];
    const float* b_hh      = (const float*)d_in[5];
    const float* w_out     = (const float*)d_in[6];
    const float* b_out     = (const float*)d_in[7];
    const float* conv_w    = (const float*)d_in[8];
    const float* conv_b    = (const float*)d_in[9];
    const float* enc_w     = (const float*)d_in[10];
    const float* enc_b     = (const float*)d_in[11];

    char* ws = (char*)d_ws;
    size_t off = 0;
    __bf16* smb    = (__bf16*)(ws + off); off = align256(off + (size_t)BATCH * SEQL * CIN * 2);
    __bf16* wih_p  = (__bf16*)(ws + off); off = align256(off + (size_t)TH3 * HID * 2);
    __bf16* whh_p  = (__bf16*)(ws + off); off = align256(off + (size_t)TH3 * HID * 2);
    __bf16* wout_p = (__bf16*)(ws + off); off = align256(off + (size_t)VOCAB * HID * 2);
    __bf16* encw_p = (__bf16*)(ws + off); off = align256(off + (size_t)HID * 32 * 2);
    __bf16* conv_p = (__bf16*)(ws + off); off = align256(off + (size_t)KCONV * OCH * 2);
    float*  h0     = (float*)(ws + off);  off = align256(off + (size_t)BATCH * HID * 4);

    float* out_lsm = (float*)d_out;
    float* out_hid = out_lsm + (size_t)BATCH * VOCAB;
    float* out_idx = out_hid + (size_t)PADLEN * BATCH * HID;

    // 1) convert sm_hidden to bf16 (flat [B, L*C])
    {
        int n = BATCH * SEQL * CIN;
        cvt_bf16_kernel<<<(n + 255) / 256, 256, 0, stream>>>(sm_hidden, smb, n);
    }
    // 2) pack weights into WMMA B-operand fragment layout
    {
        int t = (TH3 / 16) * (HID / 32) * 32;
        pack_weight_b<<<(t + 255) / 256, 256, 0, stream>>>(w_ih, wih_p, TH3, HID);
        pack_weight_b<<<(t + 255) / 256, 256, 0, stream>>>(w_hh, whh_p, TH3, HID);
    }
    {
        int t = (VOCAB / 16) * (HID / 32) * 32;
        pack_weight_b<<<(t + 255) / 256, 256, 0, stream>>>(w_out, wout_p, VOCAB, HID);
    }
    {
        int t = (HID / 16) * (32 / 32) * 32;
        pack_weight_b<<<(t + 255) / 256, 256, 0, stream>>>(enc_w, encw_p, HID, 32);
    }
    {
        int t = (OCH / 16) * NKT_C * 32;
        pack_conv_b<<<(t + 255) / 256, 256, 0, stream>>>(conv_w, conv_p);
    }
    // 3) encoder -> h0
    encoder_kernel<<<BATCH / 16, 256, 0, stream>>>(smb, conv_p, conv_b, encw_p, enc_b, h0);
    // 4) persistent GRU greedy decode (30 steps)
    decoder_kernel<<<BATCH / 16, 256, 0, stream>>>(embedding, wih_p, whh_p, wout_p,
                                                   b_ih, b_hh, b_out, h0,
                                                   out_lsm, out_hid, out_idx);
    (void)in_sizes; (void)n_in; (void)out_size; (void)ws_size;
}
